// MoELayer_730144440684
// MI455X (gfx1250) — compile-verified, matrix operand order checked by reading, not run on hardware
//
#include <hip/hip_runtime.h>
#include <hip/hip_bf16.h>

// ---------------- problem constants ----------------
constexpr int NB = 4, NT = 2048, ND = 1024, NI = 2816;
constexpr int NE = 9, NK = 2;
constexpr int NTOK = NB * NT;                 // 8192 tokens
constexpr int MTILE = 64;                     // rows per block (4 WMMA m-tiles)
constexpr int MAX_SLOTS = 17024;              // NTOK*NK + per-expert pad(<=63), 64-aligned
constexpr int TILES_M = MAX_SLOTS / MTILE;    // 266

// ---------------- WMMA types ----------------
typedef __attribute__((ext_vector_type(16))) __bf16  v16bf;
typedef __attribute__((ext_vector_type(8)))  float   v8f;
typedef __attribute__((ext_vector_type(4)))  unsigned int v4u;

union B16x16 { v16bf bf; v4u q[2]; };         // 32 bytes = 16 bf16

__device__ inline unsigned short f2bf(float f) {
    unsigned u = __float_as_uint(f);
    u += 0x7FFFu + ((u >> 16) & 1u);          // round-to-nearest-even
    return (unsigned short)(u >> 16);
}

// ---------------- init: zero out, counts; poison slot lists ----------------
__global__ void moe_init_kernel(float* out, int* counts, int* slot_token, float* slot_w) {
    size_t i = (size_t)blockIdx.x * blockDim.x + threadIdx.x;
    size_t stride = (size_t)gridDim.x * blockDim.x;
    const size_t nOut = (size_t)NTOK * ND;
    for (size_t j = i; j < nOut; j += stride) out[j] = 0.0f;
    for (size_t j = i; j < MAX_SLOTS; j += stride) { slot_token[j] = -1; slot_w[j] = 0.0f; }
    if (i < NE) counts[i] = 0;
}

// ---------------- fp32 -> bf16 conversion ----------------
__global__ void cvt_f32_bf16_kernel(const float* __restrict__ src, unsigned short* __restrict__ dst, size_t n) {
    size_t i = (size_t)blockIdx.x * blockDim.x + threadIdx.x;
    size_t stride = (size_t)gridDim.x * blockDim.x;
    for (; i < n; i += stride) dst[i] = f2bf(src[i]);
}

// ---------------- router: one wave (32 lanes) per token ----------------
__global__ void router_kernel(const float* __restrict__ x, const float* __restrict__ rw,
                              int* __restrict__ topi, float* __restrict__ topw,
                              int* __restrict__ counts) {
    const int wave = threadIdx.x >> 5;
    const int lane = threadIdx.x & 31;
    const int t = blockIdx.x * (blockDim.x >> 5) + wave;
    if (t >= NTOK) return;

    float acc[NE];
#pragma unroll
    for (int e = 0; e < NE; ++e) acc[e] = 0.0f;
    const float* xr = x + (size_t)t * ND;
    for (int d = lane; d < ND; d += 32) {
        float xv = xr[d];
#pragma unroll
        for (int e = 0; e < NE; ++e) acc[e] = fmaf(xv, rw[e * ND + d], acc[e]);
    }
#pragma unroll
    for (int e = 0; e < NE; ++e)
        for (int off = 16; off > 0; off >>= 1) acc[e] += __shfl_xor(acc[e], off, 32);

    if (lane == 0) {
        float m = acc[0];
        for (int e = 1; e < NE; ++e) m = fmaxf(m, acc[e]);
        float p[NE], s = 0.0f;
        for (int e = 0; e < NE; ++e) { p[e] = __expf(acc[e] - m); s += p[e]; }
        for (int e = 0; e < NE; ++e) p[e] /= s;
        int i0 = 0;
        for (int e = 1; e < NE; ++e) if (p[e] > p[i0]) i0 = e;
        int i1 = (i0 == 0) ? 1 : 0;
        for (int e = 0; e < NE; ++e) if (e != i0 && p[e] > p[i1]) i1 = e;
        // reference renormalizes with a softmax over the top-k *probabilities*
        float e1 = __expf(p[i1] - p[i0]);
        float w0 = 1.0f / (1.0f + e1), w1 = e1 / (1.0f + e1);
        topi[t * NK + 0] = i0; topi[t * NK + 1] = i1;
        topw[t * NK + 0] = w0; topw[t * NK + 1] = w1;
        atomicAdd(&counts[i0], 1);
        atomicAdd(&counts[i1], 1);
    }
}

// ---------------- padded prefix offsets (single thread) ----------------
__global__ void offsets_kernel(const int* __restrict__ counts, int* __restrict__ po, int* __restrict__ fill) {
    if (threadIdx.x != 0 || blockIdx.x != 0) return;
    int run = 0;
    po[0] = 0;
    for (int e = 0; e < NE; ++e) {
        fill[e] = 0;
        run += (counts[e] + (MTILE - 1)) & ~(MTILE - 1);   // pad each expert group to a 64-row tile
        po[e + 1] = run;
    }
}

// ---------------- scatter tokens into expert-grouped slots ----------------
__global__ void scatter_kernel(const int* __restrict__ topi, const float* __restrict__ topw,
                               const int* __restrict__ po, int* __restrict__ fill,
                               int* __restrict__ slot_token, float* __restrict__ slot_w) {
    int t = blockIdx.x * blockDim.x + threadIdx.x;
    if (t >= NTOK) return;
#pragma unroll
    for (int k = 0; k < NK; ++k) {
        int e = topi[t * NK + k];
        int j = atomicAdd(&fill[e], 1);
        int s = po[e] + j;
        slot_token[s] = t;
        slot_w[s] = topw[t * NK + k];
    }
}

// ---------------- GEMM1: h = silu(x Wg^T) * (x Wu^T) ----------------
// grid: (TILES_M, NI/128), block: 256 (8 waves). M=64 per block: each wave owns 16 I-columns
// and accumulates 4 m-tiles, reusing each B fragment 4x (8 WMMAs per B-pair load).
// A fragments preloaded into independent registers so ds loads overlap the WMMA pipe.
__global__ void __launch_bounds__(256)
gemm_gate_up_kernel(const unsigned short* __restrict__ xb,
                    const unsigned short* __restrict__ wgb,
                    const unsigned short* __restrict__ wub,
                    unsigned short* __restrict__ h,
                    const int* __restrict__ slot_token,
                    const int* __restrict__ po) {
    const int total = po[NE];
    const int row0 = blockIdx.x * MTILE;
    if (row0 >= total) return;
    int e = 0;
    while (e < NE - 1 && row0 >= po[e + 1]) ++e;

    const int ibase = blockIdx.y * 128;
    const int lane = threadIdx.x & 31;
    const int wave = threadIdx.x >> 5;
    const int col  = lane & 15;               // A row-in-tile == B column
    const int koff = (lane < 16) ? 0 : 8;

    // stage the 64-token x tile (64 x 1024 bf16 = 128 KB) in LDS
    __shared__ alignas(16) unsigned short xs[MTILE * ND];
    for (int i = threadIdx.x; i < MTILE * (ND / 8); i += 256) {
        int r = i >> 7;                       // row 0..63
        int c = i & 127;                      // 16-byte chunk
        int tok = slot_token[row0 + r];
        v4u val = (v4u){0u, 0u, 0u, 0u};
        if (tok >= 0) val = *(const v4u*)(xb + (size_t)tok * ND + c * 8);
        *(v4u*)(&xs[r * ND + c * 8]) = val;
    }
    __syncthreads();

    v8f cg[4] = {v8f{}, v8f{}, v8f{}, v8f{}};
    v8f cu[4] = {v8f{}, v8f{}, v8f{}, v8f{}};
    const unsigned short* wg_base = wgb + (size_t)e * NI * ND + (size_t)(ibase + wave * 16 + col) * ND;
    const unsigned short* wu_base = wub + (size_t)e * NI * ND + (size_t)(ibase + wave * 16 + col) * ND;
    const unsigned short* xrow = &xs[col * ND];

#pragma unroll 2
    for (int k0 = 0; k0 < ND; k0 += 32) {
        // B fragments (global, L2-resident weights)
        B16x16 bg, bu;
        bg.q[0] = *(const v4u*)(wg_base + k0 + koff);
        bg.q[1] = *(const v4u*)(wg_base + k0 + 16 + koff);
        bu.q[0] = *(const v4u*)(wu_base + k0 + koff);
        bu.q[1] = *(const v4u*)(wu_base + k0 + 16 + koff);
        if (k0 + 128 < ND) {
            __builtin_prefetch(wg_base + k0 + 128, 0, 1);
            __builtin_prefetch(wu_base + k0 + 128, 0, 1);
        }
        // all 4 A fragments first (independent regs -> ds loads overlap WMMA issue)
        B16x16 a[4];
#pragma unroll
        for (int mt = 0; mt < 4; ++mt) {
            a[mt].q[0] = *(const v4u*)(xrow + mt * 16 * ND + k0 + koff);
            a[mt].q[1] = *(const v4u*)(xrow + mt * 16 * ND + k0 + 16 + koff);
        }
#pragma unroll
        for (int mt = 0; mt < 4; ++mt) {
            cg[mt] = __builtin_amdgcn_wmma_f32_16x16x32_bf16(false, a[mt].bf, false, bg.bf, (short)0, cg[mt], false, false);
            cu[mt] = __builtin_amdgcn_wmma_f32_16x16x32_bf16(false, a[mt].bf, false, bu.bf, (short)0, cu[mt], false, false);
        }
    }

    // epilogue: silu(g)*u -> bf16 h; C layout: lane<16 -> M=r, lane>=16 -> M=r+8, N=lane&15
    const int mbase = (lane < 16) ? 0 : 8;
    const int icol = ibase + wave * 16 + col;
#pragma unroll
    for (int mt = 0; mt < 4; ++mt) {
#pragma unroll
        for (int r = 0; r < 8; ++r) {
            int m = mt * 16 + mbase + r;
            float g = cg[mt][r], u = cu[mt][r];
            float hv = (g / (1.0f + __expf(-g))) * u;
            h[(size_t)(row0 + m) * NI + icol] = f2bf(hv);
        }
    }
}

// ---------------- GEMM2: out[token] += w * (h Wd^T) ----------------
// grid: (TILES_M, ND/128), block: 256. Same 4x M-reuse of each B fragment; A rows are
// identical across the 8 waves -> WGP$ broadcast hits.
__global__ void __launch_bounds__(256)
gemm_down_kernel(const unsigned short* __restrict__ h,
                 const unsigned short* __restrict__ wdb,
                 const int* __restrict__ slot_token,
                 const float* __restrict__ slot_w,
                 const int* __restrict__ po,
                 float* __restrict__ out) {
    const int total = po[NE];
    const int row0 = blockIdx.x * MTILE;
    if (row0 >= total) return;
    int e = 0;
    while (e < NE - 1 && row0 >= po[e + 1]) ++e;

    const int nbase = blockIdx.y * 128;
    const int lane = threadIdx.x & 31;
    const int wave = threadIdx.x >> 5;
    const int col  = lane & 15;
    const int koff = (lane < 16) ? 0 : 8;

    v8f c[4] = {v8f{}, v8f{}, v8f{}, v8f{}};
    const unsigned short* arow = h + (size_t)(row0 + col) * NI;
    const unsigned short* brow = wdb + (size_t)e * ND * NI + (size_t)(nbase + wave * 16 + col) * NI;

#pragma unroll 2
    for (int k0 = 0; k0 < NI; k0 += 32) {
        B16x16 b;
        b.q[0] = *(const v4u*)(brow + k0 + koff);
        b.q[1] = *(const v4u*)(brow + k0 + 16 + koff);
        if (k0 + 128 < NI) __builtin_prefetch(brow + k0 + 128, 0, 1);
        B16x16 a[4];
#pragma unroll
        for (int mt = 0; mt < 4; ++mt) {
            a[mt].q[0] = *(const v4u*)(arow + (size_t)mt * 16 * NI + k0 + koff);
            a[mt].q[1] = *(const v4u*)(arow + (size_t)mt * 16 * NI + k0 + 16 + koff);
        }
#pragma unroll
        for (int mt = 0; mt < 4; ++mt) {
            c[mt] = __builtin_amdgcn_wmma_f32_16x16x32_bf16(false, a[mt].bf, false, b.bf, (short)0, c[mt], false, false);
        }
    }

    const int mbase = (lane < 16) ? 0 : 8;
    const int ocol = nbase + wave * 16 + col;
#pragma unroll
    for (int mt = 0; mt < 4; ++mt) {
#pragma unroll
        for (int r = 0; r < 8; ++r) {
            int m = mt * 16 + mbase + r;
            int tok = slot_token[row0 + m];
            if (tok >= 0) {
                float w = slot_w[row0 + m];
                atomicAdd(&out[(size_t)tok * ND + ocol], w * c[mt][r]);
            }
        }
    }
}

// ---------------- load-balancing loss ----------------
__global__ void lbloss_kernel(const int* __restrict__ counts, float* __restrict__ out) {
    if (threadIdx.x != 0 || blockIdx.x != 0) return;
    const float target = (float)(NTOK * NK) / (float)NE;
    float s = 0.0f;
    for (int e = 0; e < NE; ++e) {
        float d = (float)counts[e] - target;
        s += d * d;
    }
    out[(size_t)NTOK * ND] = s / ((float)NE * target * target);
}

// ---------------- host launcher ----------------
extern "C" void kernel_launch(void* const* d_in, const int* in_sizes, int n_in,
                              void* d_out, int out_size, void* d_ws, size_t ws_size,
                              hipStream_t stream) {
    const float* x  = (const float*)d_in[0];   // [B,T,D]
    const float* rw = (const float*)d_in[1];   // [E,D]
    const float* wg = (const float*)d_in[2];   // [E,I,D]
    const float* wu = (const float*)d_in[3];   // [E,I,D]
    const float* wd = (const float*)d_in[4];   // [E,D,I]
    float* out = (float*)d_out;                // [B*T*D] + 1 (lb_loss)

    char* ws = (char*)d_ws;
    size_t off = 0;
    auto take = [&](size_t bytes) -> char* {
        char* p = ws + off;
        off = (off + bytes + 255) & ~(size_t)255;
        return p;
    };
    unsigned short* xb  = (unsigned short*)take((size_t)NTOK * ND * 2);
    unsigned short* wgb = (unsigned short*)take((size_t)NE * NI * ND * 2);
    unsigned short* wub = (unsigned short*)take((size_t)NE * NI * ND * 2);
    unsigned short* wdb = (unsigned short*)take((size_t)NE * ND * NI * 2);
    unsigned short* hbf = (unsigned short*)take((size_t)MAX_SLOTS * NI * 2);
    int*   topi       = (int*)take((size_t)NTOK * NK * 4);
    float* topw       = (float*)take((size_t)NTOK * NK * 4);
    int*   counts     = (int*)take(NE * 4);
    int*   po         = (int*)take((NE + 1) * 4);
    int*   fill       = (int*)take(NE * 4);
    int*   slot_token = (int*)take(MAX_SLOTS * 4);
    float* slot_w     = (float*)take(MAX_SLOTS * 4);

    // 1. init (zero out/counts, poison slots)
    moe_init_kernel<<<dim3(8192), dim3(256), 0, stream>>>(out, counts, slot_token, slot_w);

    // 2. fp32 -> bf16 (weights are L2-resident afterwards: 156 MB < 192 MB L2)
    cvt_f32_bf16_kernel<<<dim3(2048), dim3(256), 0, stream>>>(x,  xb,  (size_t)NTOK * ND);
    cvt_f32_bf16_kernel<<<dim3(4096), dim3(256), 0, stream>>>(wg, wgb, (size_t)NE * NI * ND);
    cvt_f32_bf16_kernel<<<dim3(4096), dim3(256), 0, stream>>>(wu, wub, (size_t)NE * NI * ND);
    cvt_f32_bf16_kernel<<<dim3(4096), dim3(256), 0, stream>>>(wd, wdb, (size_t)NE * ND * NI);

    // 3. router (wave32 per token)
    router_kernel<<<dim3(NTOK / 8), dim3(256), 0, stream>>>(x, rw, topi, topw, counts);

    // 4. padded offsets + scatter into expert-grouped slots
    offsets_kernel<<<dim3(1), dim3(32), 0, stream>>>(counts, po, fill);
    scatter_kernel<<<dim3(NTOK / 256), dim3(256), 0, stream>>>(topi, topw, po, fill, slot_token, slot_w);

    // 5. gate/up GEMM + SiLU  (M=64 slot tiles, 128 I-columns per block)
    gemm_gate_up_kernel<<<dim3(TILES_M, NI / 128), dim3(256), 0, stream>>>(
        xb, wgb, wub, hbf, slot_token, po);

    // 6. down GEMM, scaled scatter-add into out
    gemm_down_kernel<<<dim3(TILES_M, ND / 128), dim3(256), 0, stream>>>(
        hbf, wdb, slot_token, slot_w, po, out);

    // 7. load-balancing loss
    lbloss_kernel<<<dim3(1), dim3(32), 0, stream>>>(counts, out);
}
